// Flow_SSM_Layer_7782480740551
// MI455X (gfx1250) — matrix operand order repaired and checked by reading, not compiled, not run on hardware
//
#include <hip/hip_runtime.h>
#include <hip/hip_bf16.h>
#include <cstdint>

// Problem constants (match reference)
#define B_  8
#define L_  4096
#define H_  256
#define P_  512
#define M_  1024
#define BL_ (B_*L_)
#define NC_ 16
#define CS_ (L_/NC_)

typedef _Float16 f16_t;
typedef __attribute__((ext_vector_type(16))) _Float16 v16h;
typedef __attribute__((ext_vector_type(8)))  float    v8f;

union Frag { uint4 q[2]; v16h v; };

// ---------------------------------------------------------------------------
// WMMA GEMM: Out[Mr,N] = A[Mr,K] * W[N,K]^T  (+bias) (+gelu) (+=accum)
// Block tile 128x128, 256 threads = 8 wave32 waves in 4x2 grid, each wave
// 32(M)x64(N) = 2x4 v_wmma_f32_16x16x32_f16 tiles.
// Double-buffered LDS staging via GLOBAL_LOAD_ASYNC_TO_LDS_B128 (ASYNCcnt):
// tile k+1 streams into the alternate buffer while tile k computes; waits are
// s_wait_asynccnt 0x4 (async loads complete in order -> oldest group landed).
// Requires Mr%128==0, N%128==0, K%32==0.
// ---------------------------------------------------------------------------
template<bool GELU, bool ACCUM, bool OUTH>
__global__ __launch_bounds__(256)
void gemm_f16_kernel(const f16_t* __restrict__ A, const f16_t* __restrict__ W,
                     const float* __restrict__ bias, void* __restrict__ Out,
                     int N, int K)
{
    __shared__ f16_t As[2][128][40];   // 80B row stride: 16B aligned, conflict-free
    __shared__ f16_t Ws[2][128][40];
    const int tid  = threadIdx.x;
    const int wave = tid >> 5, lane = tid & 31;
    const int wm   = wave >> 1, wn = wave & 1;
    const int lrow = lane & 15, lhalf = lane >> 4;
    const int tileM = blockIdx.y * 128, tileN = blockIdx.x * 128;

    // Per-thread staging chunk coordinates (128x32 f16 tile = 512 16B chunks)
    const int srow0 = tid >> 2, scol = (tid & 3) << 3;   // chunks 0..255
    const int srow1 = srow0 + 64;                        // chunks 256..511
    const unsigned ldsA0 = (unsigned)(size_t)&As[0][srow0][scol];
    const unsigned ldsA1 = (unsigned)(size_t)&As[0][srow1][scol];
    const unsigned ldsW0 = (unsigned)(size_t)&Ws[0][srow0][scol];
    const unsigned ldsW1 = (unsigned)(size_t)&Ws[0][srow1][scol];
    const unsigned bstride = 128u * 40u * 2u;            // 10240 B per buffer

    const f16_t* gA0 = A + (size_t)(tileM + srow0) * K + scol;
    const f16_t* gA1 = A + (size_t)(tileM + srow1) * K + scol;
    const f16_t* gW0 = W + (size_t)(tileN + srow0) * K + scol;
    const f16_t* gW1 = W + (size_t)(tileN + srow1) * K + scol;

    v8f acc[2][4] = {};

    auto stage = [&](int t) {
        unsigned boff = (t & 1) ? bstride : 0u;
        int k0 = t << 5;
        asm volatile("global_load_async_to_lds_b128 %0, %1, off"
                     :: "v"(ldsA0 + boff), "v"((unsigned long long)(gA0 + k0)) : "memory");
        asm volatile("global_load_async_to_lds_b128 %0, %1, off"
                     :: "v"(ldsA1 + boff), "v"((unsigned long long)(gA1 + k0)) : "memory");
        asm volatile("global_load_async_to_lds_b128 %0, %1, off"
                     :: "v"(ldsW0 + boff), "v"((unsigned long long)(gW0 + k0)) : "memory");
        asm volatile("global_load_async_to_lds_b128 %0, %1, off"
                     :: "v"(ldsW1 + boff), "v"((unsigned long long)(gW1 + k0)) : "memory");
    };

    auto compute = [&](int buf) {
        Frag af[2], bf[4];
        #pragma unroll
        for (int i = 0; i < 2; ++i) {
            // A 16x32 f16 layout: lanes0-15 K{0..7,16..23}, lanes16-31 K{8..15,24..31}
            int mr = wm * 32 + i * 16 + lrow;
            af[i].q[0] = *reinterpret_cast<const uint4*>(&As[buf][mr][lhalf * 8]);
            af[i].q[1] = *reinterpret_cast<const uint4*>(&As[buf][mr][lhalf * 8 + 16]);
        }
        #pragma unroll
        for (int j = 0; j < 4; ++j) {
            // B 32x16 f16 layout: lanes0-15 K{0..7,16..23}, lanes16-31 K{8..15,24..31}
            int nr = wn * 64 + j * 16 + lrow;
            bf[j].q[0] = *reinterpret_cast<const uint4*>(&Ws[buf][nr][lhalf * 8]);
            bf[j].q[1] = *reinterpret_cast<const uint4*>(&Ws[buf][nr][lhalf * 8 + 16]);
        }
        #pragma unroll
        for (int i = 0; i < 2; ++i)
            #pragma unroll
            for (int j = 0; j < 4; ++j)
                acc[i][j] = __builtin_amdgcn_wmma_f32_16x16x32_f16(
                    false, af[i].v, false, bf[j].v, (short)0, acc[i][j], false, false);
    };

    const int T = K >> 5;
    stage(0);
    int t = 0;
    for (; t + 1 < T; ++t) {
        stage(t + 1);                          // stream next tile into other buffer
        if (t + 2 < T) {                       // warm L2 for the tile after that
            __builtin_prefetch(gA0 + ((t + 2) << 5), 0, 1);
            __builtin_prefetch(gW0 + ((t + 2) << 5), 0, 1);
        }
        asm volatile("s_wait_asynccnt 0x4" ::: "memory");  // current tile landed
        __syncthreads();
        compute(t & 1);
        __syncthreads();                       // WAR guard before next stage()
    }
    asm volatile("s_wait_asynccnt 0x0" ::: "memory");      // last tile landed
    __syncthreads();
    compute(t & 1);

    // Epilogue. D layout: VGPR e, lanes0-15 -> (M=e, N=lane); lanes16-31 -> (M=e+8)
    #pragma unroll
    for (int i = 0; i < 2; ++i) {
        int rbase = tileM + wm * 32 + i * 16 + lhalf * 8;
        #pragma unroll
        for (int j = 0; j < 4; ++j) {
            int col = tileN + wn * 64 + j * 16 + lrow;
            float bv = bias ? bias[col] : 0.0f;
            #pragma unroll
            for (int e = 0; e < 8; ++e) {
                float v = acc[i][j][e] + bv;
                if (GELU) v = 0.5f * v * (1.0f + erff(v * 0.70710678118654752f));
                size_t o = (size_t)(rbase + e) * N + col;
                if (OUTH) ((f16_t*)Out)[o] = (f16_t)v;
                else {
                    float* Of = (float*)Out;
                    if (ACCUM) Of[o] += v; else Of[o] = v;
                }
            }
        }
    }
}

// ---------------------------------------------------------------------------
// adaLN: mods[b, j] = silu(cond[b,:]) . w_ada[j,:] + b_ada[j]   (j in [0,12H))
// ---------------------------------------------------------------------------
__global__ void ada_kernel(const float* __restrict__ cond, const float* __restrict__ w,
                           const float* __restrict__ bias, float* __restrict__ mods)
{
    __shared__ float sc[H_];
    int b = blockIdx.y;
    int j = blockIdx.x * 256 + threadIdx.x;
    float cv = cond[b * H_ + threadIdx.x];
    sc[threadIdx.x] = cv / (1.0f + expf(-cv));
    __syncthreads();
    float acc = 0.0f;
    const float* wr = w + (size_t)j * H_;
    for (int h = 0; h < H_; ++h) acc += wr[h] * sc[h];
    mods[(size_t)b * 12 * H_ + j] = acc + bias[j];
}

// ---------------------------------------------------------------------------
// Transpose [B,H,L] -> [B,L,H]  (32x32 LDS tiles)
// ---------------------------------------------------------------------------
__global__ void tr_in_kernel(const float* __restrict__ in, float* __restrict__ out)
{
    __shared__ float t[32][33];
    int bz = blockIdx.z, h0 = blockIdx.y * 32, l0 = blockIdx.x * 32;
    int tx = threadIdx.x, ty = threadIdx.y;
    for (int i = 0; i < 32; i += 8)
        t[ty + i][tx] = in[((size_t)bz * H_ + h0 + ty + i) * L_ + l0 + tx];
    __syncthreads();
    for (int i = 0; i < 32; i += 8)
        out[((size_t)bz * L_ + l0 + ty + i) * H_ + h0 + tx] = t[tx][ty + i];
}

// out[b,h,l] = res[b,l,h] + g[b,h] * o[b,l,h]   (fused gate+residual+transpose)
__global__ void grt_kernel(const float* __restrict__ res, const float* __restrict__ o,
                           const float* __restrict__ mods, int gIdx, float* __restrict__ out)
{
    __shared__ float t[32][33];
    int bz = blockIdx.z, l0 = blockIdx.x * 32, h0 = blockIdx.y * 32;
    int tx = threadIdx.x, ty = threadIdx.y;
    const float* mb = mods + (size_t)bz * 12 * H_ + (size_t)gIdx * H_;
    for (int i = 0; i < 32; i += 8) {
        size_t idx = ((size_t)bz * L_ + l0 + ty + i) * H_ + h0 + tx;
        t[ty + i][tx] = res[idx] + mb[h0 + tx] * o[idx];
    }
    __syncthreads();
    for (int i = 0; i < 32; i += 8)
        out[((size_t)bz * H_ + h0 + ty + i) * L_ + l0 + tx] = t[tx][ty + i];
}

// ---------------------------------------------------------------------------
// LayerNorm(+optional add) over H, then adaLN modulate, output f16
// ---------------------------------------------------------------------------
__global__ void ln_mod_kernel(const float* __restrict__ X, const float* __restrict__ X2,
                              const float* __restrict__ mods, int shIdx, int scIdx,
                              f16_t* __restrict__ U)
{
    int row = blockIdx.x;            // 0..BL-1
    int h = threadIdx.x;             // 0..255
    int b = row >> 12;               // row / L_
    size_t idx = (size_t)row * H_ + h;
    float x = X[idx];
    if (X2) x += X2[idx];
    float s = x, ss = x * x;
    for (int off = 16; off > 0; off >>= 1) {
        s  += __shfl_down(s, off);
        ss += __shfl_down(ss, off);
    }
    __shared__ float ps[8], pss[8], mv[2];
    int wv = h >> 5, ln = h & 31;
    if (ln == 0) { ps[wv] = s; pss[wv] = ss; }
    __syncthreads();
    if (h == 0) {
        float t1 = 0.f, t2 = 0.f;
        for (int w = 0; w < 8; ++w) { t1 += ps[w]; t2 += pss[w]; }
        float m = t1 * (1.0f / 256.0f);
        float v = t2 * (1.0f / 256.0f) - m * m;
        mv[0] = m; mv[1] = rsqrtf(v + 1e-6f);
    }
    __syncthreads();
    float un = (x - mv[0]) * mv[1];
    const float* mb = mods + (size_t)b * 12 * H_;
    float val = un * (1.0f + mb[(size_t)scIdx * H_ + h]) + mb[(size_t)shIdx * H_ + h];
    U[idx] = (f16_t)val;
}

// ---------------------------------------------------------------------------
// S5 discretization
// ---------------------------------------------------------------------------
__global__ void lbar_kernel(const float* __restrict__ Lre, const float* __restrict__ Lim,
                            const float* __restrict__ lstep,
                            float* lbr, float* lbi, float* lsr, float* lsi)
{
    int p = blockIdx.x * 256 + threadIdx.x;
    if (p >= P_) return;
    float dt = expf(lstep[p]);
    float ar = Lre[p] * dt, ai = Lim[p] * dt;
    float e = expf(ar);
    lbr[p] = e * cosf(ai);
    lbi[p] = e * sinf(ai);
    float eS = expf(ar * (float)CS_);             // Lbar^CS = exp(Lam*dt*CS)
    lsr[p] = eS * cosf(ai * (float)CS_);
    lsi[p] = eS * sinf(ai * (float)CS_);
}

__global__ void disc_kernel(const float* __restrict__ Lre, const float* __restrict__ Lim,
                            const float* __restrict__ lstep,
                            const float* __restrict__ Bre, const float* __restrict__ Bim,
                            f16_t* __restrict__ BbR, f16_t* __restrict__ BbI)
{
    int idx = blockIdx.x * 256 + threadIdx.x;    // P_*H_
    int p = idx >> 8;
    float dt = expf(lstep[p]);
    float lr = Lre[p], li = Lim[p];
    float e = expf(lr * dt);
    float cbr = e * cosf(li * dt) - 1.0f;        // Lbar - 1
    float cbi = e * sinf(li * dt);
    float den = lr * lr + li * li;
    float fr = (cbr * lr + cbi * li) / den;      // (Lbar-1)/Lam
    float fi = (cbi * lr - cbr * li) / den;
    float br = Bre[idx], bi = Bim[idx];
    BbR[idx] = (f16_t)(fr * br - fi * bi);
    BbI[idx] = (f16_t)(fr * bi + fi * br);
}

// ---------------------------------------------------------------------------
// Chunked linear scan with constant complex coefficient Lbar_p.
// Bu is stored f16 (it is already the product of two f16 operands); scan
// state stays f32. Pass 1: per-(b,p,chunk) local final state. Pass 2:
// combine carries via Lbar^CS and rescan, writing f16 xs into [BL, 2P]
// (fwd cols [0,P), bwd [P,2P)). Grid: B*P*NC = 65536 threads.
// ---------------------------------------------------------------------------
__global__ void scan_pass1(const f16_t* __restrict__ BuR, const f16_t* __restrict__ BuI,
                           const float* __restrict__ lbr, const float* __restrict__ lbi,
                           float2* __restrict__ F, int back)
{
    int idx = blockIdx.x * 256 + threadIdx.x;
    int p = idx & (P_ - 1);
    int c = (idx >> 9) & (NC_ - 1);
    int b = idx >> 13;
    float ar = lbr[p], ai = lbi[p];
    float xr = 0.f, xi = 0.f;
    for (int s = 0; s < CS_; ++s) {
        int k = c * CS_ + s;
        int l = back ? (L_ - 1 - k) : k;
        size_t o = ((size_t)b * L_ + l) * P_ + p;
        float br = (float)BuR[o], bi = (float)BuI[o];
        float nr = ar * xr - ai * xi + br;
        float ni = ar * xi + ai * xr + bi;
        xr = nr; xi = ni;
    }
    F[((size_t)b * NC_ + c) * P_ + p] = make_float2(xr, xi);
}

__global__ void scan_pass2(const f16_t* __restrict__ BuR, const f16_t* __restrict__ BuI,
                           const float* __restrict__ lbr, const float* __restrict__ lbi,
                           const float* __restrict__ lsr, const float* __restrict__ lsi,
                           const float2* __restrict__ F,
                           f16_t* __restrict__ XsR, f16_t* __restrict__ XsI, int back)
{
    int idx = blockIdx.x * 256 + threadIdx.x;
    int p = idx & (P_ - 1);
    int c = (idx >> 9) & (NC_ - 1);
    int b = idx >> 13;
    float sr = lsr[p], si = lsi[p];
    float cr = 0.f, ci = 0.f;
    for (int j = 0; j < c; ++j) {                 // carry = Lbar^CS*carry + F_j
        float2 Fv = F[((size_t)b * NC_ + j) * P_ + p];
        float tr = sr * cr - si * ci + Fv.x;
        float ti = sr * ci + si * cr + Fv.y;
        cr = tr; ci = ti;
    }
    float ar = lbr[p], ai = lbi[p];
    float xr = cr, xi = ci;
    int colOff = back ? P_ : 0;
    for (int s = 0; s < CS_; ++s) {
        int k = c * CS_ + s;
        int l = back ? (L_ - 1 - k) : k;
        size_t o = ((size_t)b * L_ + l) * P_ + p;
        float nr = ar * xr - ai * xi + (float)BuR[o];
        float ni = ar * xi + ai * xr + (float)BuI[o];
        xr = nr; xi = ni;
        size_t xo = ((size_t)b * L_ + l) * (2 * P_) + colOff + p;
        XsR[xo] = (f16_t)xr;
        XsI[xo] = (f16_t)xi;
    }
}

// ---------------------------------------------------------------------------
// S5 epilogue: cond = g * (2*Y + D*u); optionally outP = res + cond
// ---------------------------------------------------------------------------
__global__ void s5_epi_kernel(const float* __restrict__ Y, const f16_t* __restrict__ U,
                              const float* __restrict__ D, const float* __restrict__ mods,
                              int gIdx, const float* __restrict__ res,
                              f16_t* __restrict__ condH, float* __restrict__ outP)
{
    size_t idx = (size_t)blockIdx.x * 256 + threadIdx.x;
    int h = (int)(idx & 255);
    int b = (int)(idx >> 20);                     // /(L_*H_)
    float u = (float)U[idx];
    float g = mods[(size_t)b * 12 * H_ + (size_t)gIdx * H_ + h];
    float cval = g * (2.0f * Y[idx] + D[h] * u);
    condH[idx] = (f16_t)cval;
    if (outP) outP[idx] = res[idx] + cval;
}

__global__ void conv_h_kernel(const float* __restrict__ in, f16_t* __restrict__ out,
                              int n, float scale)
{
    int i = blockIdx.x * 256 + threadIdx.x;
    if (i < n) out[i] = (f16_t)(scale * in[i]);
}

// ---------------------------------------------------------------------------
// Host orchestration
// ---------------------------------------------------------------------------
static void launch_gemm(int mode, const f16_t* A, const f16_t* W, const float* bias,
                        void* Out, int Mr, int N, int K, hipStream_t s)
{
    dim3 g(N / 128, Mr / 128), b(256);
    if (mode == 0)      gemm_f16_kernel<false, false, false><<<g, b, 0, s>>>(A, W, bias, Out, N, K);
    else if (mode == 1) gemm_f16_kernel<false, true,  false><<<g, b, 0, s>>>(A, W, bias, Out, N, K);
    else if (mode == 2) gemm_f16_kernel<true,  false, true ><<<g, b, 0, s>>>(A, W, bias, Out, N, K);
    else                gemm_f16_kernel<false, false, true ><<<g, b, 0, s>>>(A, W, bias, Out, N, K);
}

extern "C" void kernel_launch(void* const* d_in, const int* in_sizes, int n_in,
                              void* d_out, int out_size, void* d_ws, size_t ws_size,
                              hipStream_t stream)
{
    (void)in_sizes; (void)n_in; (void)out_size; (void)ws_size;

    const float* ppg  = (const float*)d_in[0];
    const float* xt   = (const float*)d_in[1];
    const float* cond = (const float*)d_in[2];
    const float* wada = (const float*)d_in[3];
    const float* bada = (const float*)d_in[4];

    // Branch params: 0 = ppg (idx 5..12), 1 = tgt (idx 13..20)
    const float *Lre[2], *Lim[2], *Bre[2], *Bim[2], *Cre[2], *Cim[2], *Dv[2], *lstep[2];
    for (int br = 0; br < 2; ++br) {
        int o = 5 + br * 8;
        Lre[br] = (const float*)d_in[o + 0]; Lim[br]  = (const float*)d_in[o + 1];
        Bre[br] = (const float*)d_in[o + 2]; Bim[br]  = (const float*)d_in[o + 3];
        Cre[br] = (const float*)d_in[o + 4]; Cim[br]  = (const float*)d_in[o + 5];
        Dv[br]  = (const float*)d_in[o + 6]; lstep[br]= (const float*)d_in[o + 7];
    }
    // MLPs: 0 pre_ppg, 1 mlp_ppg, 2 pre_tgt, 3 post_tgt, 4 mlp_tgt
    const float *w1f[5], *b1f[5], *w2f[5], *b2f[5];
    for (int m = 0; m < 5; ++m) {
        int o = 21 + m * 4;
        w1f[m] = (const float*)d_in[o + 0]; b1f[m] = (const float*)d_in[o + 1];
        w2f[m] = (const float*)d_in[o + 2]; b2f[m] = (const float*)d_in[o + 3];
    }

    // ---- workspace carve ----
    char* base = (char*)d_ws;
    size_t off = 0;
    auto carve = [&](size_t bytes) -> char* {
        char* p = base + off;
        off += (bytes + 255) & ~(size_t)255;
        return p;
    };
    float*  mods = (float*)carve((size_t)B_ * 12 * H_ * 4);
    float*  lb[2][4];
    f16_t*  Bb[2][2];
    f16_t*  Ch[2][2];
    for (int br = 0; br < 2; ++br) {
        for (int k = 0; k < 4; ++k) lb[br][k] = (float*)carve(P_ * 4);
        Bb[br][0] = (f16_t*)carve((size_t)P_ * H_ * 2);
        Bb[br][1] = (f16_t*)carve((size_t)P_ * H_ * 2);
        Ch[br][0] = (f16_t*)carve((size_t)H_ * 2 * P_ * 2);
        Ch[br][1] = (f16_t*)carve((size_t)H_ * 2 * P_ * 2);
    }
    f16_t *w1h[5], *w2h[5];
    for (int m = 0; m < 5; ++m) {
        w1h[m] = (f16_t*)carve((size_t)M_ * H_ * 2);
        w2h[m] = (f16_t*)carve((size_t)H_ * M_ * 2);
    }
    float2* Fbuf = (float2*)carve((size_t)B_ * P_ * NC_ * 8);

    float* resP = (float*)carve((size_t)BL_ * H_ * 4);
    float* resT = (float*)carve((size_t)BL_ * H_ * 4);
    float* Ybuf = (float*)carve((size_t)BL_ * H_ * 4);
    float* P1   = (float*)carve((size_t)BL_ * H_ * 4);
    float* ACC  = (float*)carve((size_t)BL_ * H_ * 4);
    f16_t* U    = (f16_t*)carve((size_t)BL_ * H_ * 2);
    f16_t* CNDH = (f16_t*)carve((size_t)BL_ * H_ * 2);
    f16_t* H1   = (f16_t*)carve((size_t)BL_ * M_ * 2);
    f16_t* BuR  = (f16_t*)carve((size_t)BL_ * P_ * 2);
    f16_t* BuI  = (f16_t*)carve((size_t)BL_ * P_ * 2);
    f16_t* XsR  = (f16_t*)carve((size_t)BL_ * 2 * P_ * 2);
    f16_t* XsI  = (f16_t*)carve((size_t)BL_ * 2 * P_ * 2);

    float* outP = (float*)d_out;
    float* outD = (float*)d_out + (size_t)B_ * H_ * L_;

    const dim3 b256(256);
    const dim3 tgrid(L_ / 32, H_ / 32, B_), tblk(32, 8);
    const int nEW = (int)(((size_t)BL_ * H_) / 256);        // 32768 blocks
    const int nW  = (M_ * H_) / 256;                        // 1024 blocks

    // 1. adaLN mods
    ada_kernel<<<dim3(12, B_), b256, 0, stream>>>(cond, wada, bada, mods);

    // 2. transpose inputs to [B,L,H]
    tr_in_kernel<<<tgrid, tblk, 0, stream>>>(ppg, resP);
    tr_in_kernel<<<tgrid, tblk, 0, stream>>>(xt, resT);

    // 3. weight conversions to f16 (fold -1 into Cim for the real-part GEMM)
    for (int m = 0; m < 5; ++m) {
        conv_h_kernel<<<nW, b256, 0, stream>>>(w1f[m], w1h[m], M_ * H_, 1.0f);
        conv_h_kernel<<<nW, b256, 0, stream>>>(w2f[m], w2h[m], H_ * M_, 1.0f);
    }
    for (int br = 0; br < 2; ++br) {
        conv_h_kernel<<<(H_ * 2 * P_) / 256, b256, 0, stream>>>(Cre[br], Ch[br][0], H_ * 2 * P_, 1.0f);
        conv_h_kernel<<<(H_ * 2 * P_) / 256, b256, 0, stream>>>(Cim[br], Ch[br][1], H_ * 2 * P_, -1.0f);
        lbar_kernel<<<2, b256, 0, stream>>>(Lre[br], Lim[br], lstep[br],
                                            lb[br][0], lb[br][1], lb[br][2], lb[br][3]);
        disc_kernel<<<(P_ * H_) / 256, b256, 0, stream>>>(Lre[br], Lim[br], lstep[br],
                                                          Bre[br], Bim[br], Bb[br][0], Bb[br][1]);
    }

    // helper: run S5 branch br on normalized/modded input U -> Ybuf (= Re part)
    auto run_s5 = [&](int br) {
        launch_gemm(3, U, Bb[br][0], nullptr, BuR, BL_, P_, H_, stream);   // f16 out
        launch_gemm(3, U, Bb[br][1], nullptr, BuI, BL_, P_, H_, stream);
        for (int back = 0; back < 2; ++back) {
            scan_pass1<<<256, b256, 0, stream>>>(BuR, BuI, lb[br][0], lb[br][1], Fbuf, back);
            scan_pass2<<<256, b256, 0, stream>>>(BuR, BuI, lb[br][0], lb[br][1],
                                                 lb[br][2], lb[br][3], Fbuf, XsR, XsI, back);
        }
        launch_gemm(0, XsR, Ch[br][0], nullptr, Ybuf, BL_, H_, 2 * P_, stream);
        launch_gemm(1, XsI, Ch[br][1], nullptr, Ybuf, BL_, H_, 2 * P_, stream);
    };

    // ---- ppg branch ----
    ln_mod_kernel<<<BL_, b256, 0, stream>>>(resP, nullptr, mods, 0, 1, U);   // sh_sp, sc_sp
    run_s5(0);
    // ppg_cond = g_sp*(2Y + D*u); P1 = resP + ppg_cond
    s5_epi_kernel<<<nEW, b256, 0, stream>>>(Ybuf, U, Dv[0], mods, 2, resP, CNDH, P1);
    // mlp_ppg on ln(P1) modded
    ln_mod_kernel<<<BL_, b256, 0, stream>>>(P1, nullptr, mods, 3, 4, U);     // sh_mp, sc_mp
    launch_gemm(2, U, w1h[1], b1f[1], H1, BL_, M_, H_, stream);
    launch_gemm(0, H1, w2h[1], b2f[1], Ybuf, BL_, H_, M_, stream);
    grt_kernel<<<tgrid, tblk, 0, stream>>>(resP, Ybuf, mods, 5, outP);       // g_mp
    // pc = mlp(ppg_cond; pre_ppg) -> ACC
    launch_gemm(2, CNDH, w1h[0], b1f[0], H1, BL_, M_, H_, stream);
    launch_gemm(0, H1, w2h[0], b2f[0], ACC, BL_, H_, M_, stream);

    // ---- tgt branch ----
    ln_mod_kernel<<<BL_, b256, 0, stream>>>(resT, nullptr, mods, 6, 7, U);   // sh_st, sc_st
    run_s5(1);
    // tc0 = g_st*(2Y + D*u) (f16 only)
    s5_epi_kernel<<<nEW, b256, 0, stream>>>(Ybuf, U, Dv[1], mods, 8, resT, CNDH, nullptr);
    // tc1 = mlp(tc0; pre_tgt) + pc   (accumulate into ACC)
    launch_gemm(2, CNDH, w1h[2], b1f[2], H1, BL_, M_, H_, stream);
    launch_gemm(1, H1, w2h[2], b2f[2], ACC, BL_, H_, M_, stream);
    // tc2 = mlp(tc1; post_tgt) -> Ybuf
    conv_h_kernel<<<nEW, b256, 0, stream>>>(ACC, U, BL_ * H_, 1.0f);
    launch_gemm(2, U, w1h[3], b1f[3], H1, BL_, M_, H_, stream);
    launch_gemm(0, H1, w2h[3], b2f[3], Ybuf, BL_, H_, M_, stream);
    // t2 = resT + tc2 fused into LN; mlp_tgt
    ln_mod_kernel<<<BL_, b256, 0, stream>>>(resT, Ybuf, mods, 9, 10, U);     // sh_mt, sc_mt
    launch_gemm(2, U, w1h[4], b1f[4], H1, BL_, M_, H_, stream);
    launch_gemm(0, H1, w2h[4], b2f[4], Ybuf, BL_, H_, M_, stream);
    grt_kernel<<<tgrid, tblk, 0, stream>>>(resT, Ybuf, mods, 11, outD);      // g_mt
}
